// FilteredLeakyReLU_73212012527678
// MI455X (gfx1250) — matrix operand order repaired and checked by reading, not compiled, not run on hardware
//
#include <hip/hip_runtime.h>

typedef __attribute__((ext_vector_type(16))) _Float16 v16h;
typedef __attribute__((ext_vector_type(8)))  _Float16 v8h;
typedef __attribute__((ext_vector_type(2)))  _Float16 v2h;
typedef __attribute__((ext_vector_type(8)))  float    v8f;

// ---------------------------------------------------------------------------
// filtered_lrelu as four banded-matrix WMMA passes.
// One wave32 handles a 2x2 quad of 16x16 output tiles (32x32 outputs) with a
// shared 48x48 x-window in LDS; per-tile pipeline (rolled loop over 4 tiles):
//   S1: Zh = X*U            (A=s_x,  B=U)   -> store Zh^T
//   S2: Z^T = Zh^T*U        (A=s_zh, B=U)   -> lrelu -> store Z
//   S3: Y1 = Z*D            (A=s_z,  B=D)   -> store Y1^T
//   S4: Out = D^T*Y1        (A=D^T,  B from Y1^T, contiguous) -> global
// Band matrices (position independent):
//   U[j][v] = 2*fu[2j+1-v], D[v][o] = fd[v-2o]
// Constant band fragments prebuilt into d_ws by a tiny setup kernel; every
// LDS fragment access is an aligned 16B b128 op.
// ---------------------------------------------------------------------------

__device__ __forceinline__ v16h cat16(v8h lo, v8h hi) {
  return __builtin_shufflevector(lo, hi, 0, 1, 2, 3, 4, 5, 6, 7,
                                 8, 9, 10, 11, 12, 13, 14, 15);
}

__device__ __forceinline__ v8f wmma32(v16h a, v16h b, v8f c) {
  return __builtin_amdgcn_wmma_f32_16x16x32_f16(
      false, a, false, b, (short)0, c, false, false);
}

// A fragment (16x32 f16), K contiguous per lane: two aligned 16B LDS loads
__device__ __forceinline__ v16h load_a(const _Float16* buf, int ld,
                                       int row0, int col0) {
  const int lane = threadIdx.x;
  const int m = lane & 15, kh = lane >> 4;
  const _Float16* p = buf + (row0 + m) * ld + col0 + 16 * kh;
  const v8h lo = *(const v8h*)p;         // K halfs [0..7]  (+16*kh)
  const v8h hi = *(const v8h*)(p + 8);   // K halfs [8..15] (+16*kh)
  return cat16(lo, hi);
}

// B fragment (32x16 f16) from a transposed buffer (row n holds K contiguous)
__device__ __forceinline__ v16h load_bT(const _Float16* buf, int ld, int col0) {
  const int lane = threadIdx.x;
  const int n = lane & 15, kh = lane >> 4;
  const _Float16* p = buf + n * ld + col0 + 16 * kh;
  const v8h lo = *(const v8h*)p;
  const v8h hi = *(const v8h*)(p + 8);
  return cat16(lo, hi);
}

// Fragment-transposed C/D store: element (M,N) -> buf[(N0+N)*ld + M0+M]
// per lane: 8 consecutive halfs = one aligned 16B LDS store
template <bool LRELU>
__device__ __forceinline__ void store_T(_Float16* buf, int ld, int M0, int N0,
                                        v8f c) {
  const int lane = threadIdx.x;
  const int n = lane & 15, kh = lane >> 4;
  v8h o;
#pragma unroll
  for (int v = 0; v < 8; ++v) {
    float t = c[v];
    if (LRELU) t = (t >= 0.0f) ? t : 0.01f * t;
    o[v] = (_Float16)t;
  }
  *(v8h*)(buf + (N0 + n) * ld + M0 + 8 * kh) = o;
}

// ---- constant band fragments (built once by setup kernel) -----------------
__device__ __forceinline__ _Float16 uval(const _Float16* fu2, int j, int v) {
  const int t = 2 * j + 1 - v;
  return (t >= 0 && t < 12) ? fu2[t] : (_Float16)0.0f;
}
__device__ __forceinline__ _Float16 dval(const _Float16* fd, int v, int o) {
  const int t = v - 2 * o;
  return (t >= 0 && t < 12) ? fd[t] : (_Float16)0.0f;
}
__device__ __forceinline__ int a_kbase(int p, int kh) {
  return 2 * p + 16 * kh;
}

__device__ __forceinline__ v16h make_b_U(const _Float16* fu2, int vcol0) {
  const int lane = threadIdx.x;
  const int n = lane & 15, kh = lane >> 4;
  v16h b;
#pragma unroll
  for (int p = 0; p < 8; ++p) {
    const int k = 2 * p + 16 * kh;
    b[2 * p]     = uval(fu2, k,     vcol0 + n);
    b[2 * p + 1] = uval(fu2, k + 1, vcol0 + n);
  }
  return b;
}
__device__ __forceinline__ v16h make_b_D(const _Float16* fd, int kc) {
  const int lane = threadIdx.x;
  const int n = lane & 15, kh = lane >> 4;
  v16h b;
#pragma unroll
  for (int p = 0; p < 8; ++p) {
    const int k = 2 * p + 16 * kh + kc * 32;
    b[2 * p]     = dval(fd, k,     n);
    b[2 * p + 1] = dval(fd, k + 1, n);
  }
  return b;
}
__device__ __forceinline__ v16h make_a_Dt(const _Float16* fd, int kc) {
  const int lane = threadIdx.x;
  const int m = lane & 15, kh = lane >> 4;
  v16h a;
#pragma unroll
  for (int p = 0; p < 8; ++p) {
    const int k = a_kbase(p, kh) + kc * 32;
    a[2 * p]     = dval(fd, k,     m);
    a[2 * p + 1] = dval(fd, k + 1, m);
  }
  return a;
}

// one wave: build the 7 constant fragments into ws in fragment-major layout
__global__ __launch_bounds__(32) void flrelu_build_frags(
    const float* __restrict__ fup, const float* __restrict__ fdn,
    _Float16* __restrict__ ws) {
  __shared__ _Float16 sfu2[12], sfd[12];
  const int lane = threadIdx.x;
  if (lane < 12)      sfu2[lane]     = (_Float16)(2.0f * fup[lane]);
  else if (lane < 24) sfd[lane - 12] = (_Float16)fdn[lane - 12];
  __syncthreads();
  v16h f[7];
  f[0] = make_b_U(sfu2, 0);
  f[1] = make_b_U(sfu2, 16);
  f[2] = make_b_U(sfu2, 32);
  f[3] = make_b_D(sfd, 0);
  f[4] = make_b_D(sfd, 1);
  f[5] = make_a_Dt(sfd, 0);
  f[6] = make_a_Dt(sfd, 1);
#pragma unroll
  for (int fi = 0; fi < 7; ++fi)
#pragma unroll
    for (int e = 0; e < 16; ++e) ws[fi * 512 + lane * 16 + e] = f[fi][e];
}

// load one prebuilt fragment: two coalesced global b128 loads per lane
__device__ __forceinline__ v16h ldfrag(const _Float16* __restrict__ ws,
                                       int fi) {
  const _Float16* p = ws + (fi << 9) + ((int)threadIdx.x << 4);
  const v8h lo = *(const v8h*)p;
  const v8h hi = *(const v8h*)(p + 8);
  return cat16(lo, hi);
}

__global__ __launch_bounds__(32) void filtered_lrelu_wmma_kernel(
    const float* __restrict__ x, const _Float16* __restrict__ ws,
    float* __restrict__ out) {
  __shared__ alignas(16) _Float16 s_x[48][56];    // shared quad x window (f16)
  __shared__ alignas(16) _Float16 s_zh[48][40];   // Zh^T: [v_c][j]
  __shared__ alignas(16) _Float16 s_z[48][72];    // Z: [v_r][v_c], K pad 64
  __shared__ alignas(16) _Float16 s_y1t[16][72];  // Y1^T: [o_c][v_r], pad 64

  const int lane = threadIdx.x;
  const int bid  = blockIdx.x;
  const int img  = bid >> 4;             // B*C = 1024 channel images
  const int quad = bid & 15;             // 4x4 quads of 32x32 outputs
  const int qy   = (quad >> 2) << 5;
  const int qx   = (quad & 3) << 5;
  const float* __restrict__ xin = x + (size_t)img * (130 * 130);
  float* __restrict__ yo        = out + (size_t)img * (128 * 128);
  const int yb = qy - 4, xb = qx - 4;    // 48x48 window covers all 4 tiles

  // constant band fragments (L2-resident, 2x b128 each; live across tile loop)
  const v16h bU0 = ldfrag(ws, 0), bU1 = ldfrag(ws, 1), bU2 = ldfrag(ws, 2);
  const v16h bD0 = ldfrag(ws, 3), bD1 = ldfrag(ws, 4);
  const v16h aDt0 = ldfrag(ws, 5), aDt1 = ldfrag(ws, 6);

  // --- stage 0: zero-padded 48x48 x window (pairs -> b32 stores), K-pad zeros
  for (int i = lane; i < 48 * 24; i += 32) {
    const int r  = i / 24;
    const int pc = i - r * 24;
    const int gy = yb + r, gx0 = xb + pc * 2;
    float a = 0.0f, b = 0.0f;
    if (gy >= 0 && gy < 130) {
      if (gx0 >= 0 && gx0 < 130)         a = xin[gy * 130 + gx0];
      if (gx0 + 1 >= 0 && gx0 + 1 < 130) b = xin[gy * 130 + gx0 + 1];
    }
    v2h p; p[0] = (_Float16)a; p[1] = (_Float16)b;
    *(v2h*)(&s_x[r][pc * 2]) = p;
  }
  {
    const v8h z = {};
#pragma unroll
    for (int i = lane; i < 96; i += 32)           // s_z cols 48..63
      *(v8h*)(&s_z[i >> 1][48 + 8 * (i & 1)]) = z;
    *(v8h*)(&s_y1t[lane >> 1][48 + 8 * (lane & 1)]) = z;  // s_y1t cols 48..63
  }
  __syncthreads();

  // --- tile loop: 2x2 tiles of 16x16, rolled (stage buffers reused) --------
  for (int t = 0; t < 4; ++t) {
    const int ty2 = t >> 1, tx2 = t & 1;
    const int xr0 = ty2 << 4;            // window row base of this tile
    const int xc0 = tx2 << 4;            // window col base of this tile

    // S1: Zh(32x48) = X*U, store Zh^T   (6 WMMA)
#pragma unroll
    for (int rb = 0; rb < 2; ++rb) {
      const v16h a = load_a(&s_x[0][0], 56, xr0 + rb * 16, xc0);
      v8f c0 = {}, c1 = {}, c2 = {};
      c0 = wmma32(a, bU0, c0);
      c1 = wmma32(a, bU1, c1);
      c2 = wmma32(a, bU2, c2);
      store_T<false>(&s_zh[0][0], 40, rb * 16, 0,  c0);
      store_T<false>(&s_zh[0][0], 40, rb * 16, 16, c1);
      store_T<false>(&s_zh[0][0], 40, rb * 16, 32, c2);
    }
    __syncthreads();

    // S2: Z^T(48x48) = Zh^T*U, lrelu, store Z   (9 WMMA)
#pragma unroll
    for (int rb = 0; rb < 3; ++rb) {  // rb indexes v_c blocks (rows of Zh^T)
      const v16h a = load_a(&s_zh[0][0], 40, rb * 16, 0);
      v8f c0 = {}, c1 = {}, c2 = {};
      c0 = wmma32(a, bU0, c0);
      c1 = wmma32(a, bU1, c1);
      c2 = wmma32(a, bU2, c2);
      store_T<true>(&s_z[0][0], 72, rb * 16, 0,  c0);
      store_T<true>(&s_z[0][0], 72, rb * 16, 16, c1);
      store_T<true>(&s_z[0][0], 72, rb * 16, 32, c2);
    }
    __syncthreads();

    // S3: Y1(48x16) = Z*D (K=64 padded), store Y1^T   (6 WMMA)
#pragma unroll
    for (int rb = 0; rb < 3; ++rb) {
      v8f c = {};
      c = wmma32(load_a(&s_z[0][0], 72, rb * 16, 0),  bD0, c);
      c = wmma32(load_a(&s_z[0][0], 72, rb * 16, 32), bD1, c);
      store_T<false>(&s_y1t[0][0], 72, rb * 16, 0, c);
    }
    __syncthreads();

    // S4: Out(16x16) = D^T*Y1 (B contiguous from Y1^T)   (2 WMMA)
    {
      v8f c = {};
      c = wmma32(aDt0, load_bT(&s_y1t[0][0], 72, 0),  c);
      c = wmma32(aDt1, load_bT(&s_y1t[0][0], 72, 32), c);
      const int oy0 = qy + xr0, ox0 = qx + xc0;
      const int n = lane & 15, kh = lane >> 4;
#pragma unroll
      for (int v = 0; v < 8; ++v)
        yo[(size_t)(oy0 + v + 8 * kh) * 128 + ox0 + n] = c[v];
    }
    __syncthreads();  // stage buffers reused by next tile
  }
}

extern "C" void kernel_launch(void* const* d_in, const int* in_sizes, int n_in,
                              void* d_out, int out_size, void* d_ws, size_t ws_size,
                              hipStream_t stream) {
  (void)in_sizes; (void)n_in; (void)out_size; (void)ws_size;
  const float* x  = (const float*)d_in[0];
  const float* fu = (const float*)d_in[1];
  const float* fd = (const float*)d_in[2];
  float* out = (float*)d_out;
  _Float16* ws = (_Float16*)d_ws;  // 7 fragments * 512 halfs = 7 KB

  flrelu_build_frags<<<dim3(1), dim3(32), 0, stream>>>(fu, fd, ws);
  // 1024 channel-images * 16 quads; one wave32 per 2x2 group of 16x16 tiles
  filtered_lrelu_wmma_kernel<<<dim3(1024 * 16), dim3(32), 0, stream>>>(x, ws, out);
}
